// RBF_Attention_52140902974027
// MI455X (gfx1250) — matrix operand order repaired and checked by reading, not compiled, not run on hardware
//
#include <hip/hip_runtime.h>
#include <math.h>

#define BB 4
#define SS 4096
#define QD 256
#define KDIM 128
#define VDIM 64

typedef __attribute__((ext_vector_type(16))) _Float16 v16h;
typedef __attribute__((ext_vector_type(8)))  float    v8f;

// ---- Kernel 1: tq = l2norm(query @ W^T), one block per (b, which) ----
__global__ void proj_norm_kernel(const float* __restrict__ q1,
                                 const float* __restrict__ q2,
                                 const float* __restrict__ W,
                                 float* __restrict__ tq /* [2][B][KD] */) {
  const int which = blockIdx.x & 1;
  const int b     = blockIdx.x >> 1;
  const int t     = threadIdx.x;             // 0..127 (one output feature each)
  const float* q    = (which == 0 ? q1 : q2) + b * QD;
  const float* wrow = W + (size_t)t * QD;
  float dot = 0.f;
  for (int d = 0; d < QD; d += 4)
    dot += q[d]*wrow[d] + q[d+1]*wrow[d+1] + q[d+2]*wrow[d+2] + q[d+3]*wrow[d+3];
  __shared__ float red[128];
  red[t] = dot * dot;
  __syncthreads();
  for (int off = 64; off > 0; off >>= 1) {
    if (t < off) red[t] += red[t + off];
    __syncthreads();
  }
  float nrm = fmaxf(sqrtf(red[0]), 1e-12f);
  tq[((size_t)which * BB + b) * KDIM + t] = dot / nrm;
}

// ---- Kernel 2: scores (normalized-key dot) + ||v||^2 per row ----
__global__ void scores_sq_kernel(const float* __restrict__ k1,
                                 const float* __restrict__ k2,
                                 const float* __restrict__ v1,
                                 const float* __restrict__ v2,
                                 const float* __restrict__ tq,   // [2][B][KD]
                                 float* __restrict__ scores1,
                                 float* __restrict__ scores2,
                                 float* __restrict__ sq1,
                                 float* __restrict__ sq2) {
  const int t = blockIdx.x * blockDim.x + threadIdx.x;  // 0..B*S-1
  const int b = t >> 12;
  const float* tqa = tq + (size_t)b * KDIM;
  const float* tqb = tq + (size_t)(BB + b) * KDIM;
  const float* kr1 = k1 + (size_t)t * KDIM;
  const float* kr2 = k2 + (size_t)t * KDIM;
  float d1 = 0.f, n1 = 0.f, d2 = 0.f, n2 = 0.f;
  for (int d = 0; d < KDIM; ++d) {
    float a = kr1[d]; d1 += a * tqa[d]; n1 += a * a;
    float c = kr2[d]; d2 += c * tqb[d]; n2 += c * c;
  }
  scores1[t] = d1 / fmaxf(sqrtf(n1), 1e-12f);
  scores2[t] = d2 / fmaxf(sqrtf(n2), 1e-12f);
  const float* vr1 = v1 + (size_t)t * VDIM;
  const float* vr2 = v2 + (size_t)t * VDIM;
  float s1 = 0.f, s2 = 0.f;
  for (int d = 0; d < VDIM; ++d) { s1 += vr1[d]*vr1[d]; s2 += vr2[d]*vr2[d]; }
  sq1[t] = s1; sq2[t] = s2;
}

// ---- Kernel 3: softmax over S, one block per (which, b); writes attn -> d_out ----
__global__ void softmax_kernel(const float* __restrict__ scores1,
                               const float* __restrict__ scores2,
                               float* __restrict__ out) {
  const int which = blockIdx.x >> 2;
  const int b     = blockIdx.x & 3;
  const float* sc = (which == 0 ? scores1 : scores2) + (size_t)b * SS;
  float* dst = out + (size_t)which * (BB * SS) + (size_t)b * SS;
  const int t = threadIdx.x;  // 256 threads
  __shared__ float red[256];
  float mx = -3.402823466e38f;
  for (int s = t; s < SS; s += 256) mx = fmaxf(mx, sc[s]);
  red[t] = mx; __syncthreads();
  for (int off = 128; off > 0; off >>= 1) {
    if (t < off) red[t] = fmaxf(red[t], red[t + off]);
    __syncthreads();
  }
  mx = red[0]; __syncthreads();
  float sum = 0.f;
  for (int s = t; s < SS; s += 256) sum += __expf(sc[s] - mx);
  red[t] = sum; __syncthreads();
  for (int off = 128; off > 0; off >>= 1) {
    if (t < off) red[t] += red[t + off];
    __syncthreads();
  }
  const float inv = 1.f / red[0];
  for (int s = t; s < SS; s += 256) dst[s] = __expf(sc[s] - mx) * inv;
}

// ---- Kernel 4: fused Gram-WMMA + RBF + weighted reduction ----
// One block per (batch, i-tile of 16 rows); 8 waves sweep 256 j-tiles.
__global__ void rbf_wmma_kernel(const float* __restrict__ v1,
                                const float* __restrict__ v2,
                                const float* __restrict__ sq1,
                                const float* __restrict__ sq2,
                                const float* __restrict__ attn,  // d_out: attn1 | attn2
                                float* __restrict__ partials) {
  const int b     = blockIdx.x >> 8;
  const int iTile = blockIdx.x & 255;
  const int i0    = iTile * 16;
  const int tid   = threadIdx.x;
  const int wave  = tid >> 5;
  const int lane  = tid & 31;
  const int half  = lane >> 4;
  const int nIdx  = lane & 15;

  const float* attn1 = attn + (size_t)b * SS;
  const float* attn2 = attn + (size_t)BB * SS + (size_t)b * SS;
  const float* sq1b  = sq1 + (size_t)b * SS;
  const float* sq2b  = sq2 + (size_t)b * SS;
  const float* v1b   = v1 + (size_t)b * SS * VDIM;
  const float* v2b   = v2 + (size_t)b * SS * VDIM;

  // A fragments (16x32 f16 each, K=0..31 and K=32..63), contiguous K order (B uses the same):
  // lane: M = lane%16 ; element e -> K = 16*half + e
  v16h a_lo, a_hi;
  {
    const float* row = v1b + (size_t)(i0 + nIdx) * VDIM;
#pragma unroll
    for (int e = 0; e < 16; ++e) {
      const int k = 16 * half + e;
      a_lo[e] = (_Float16)row[k];
      a_hi[e] = (_Float16)row[k + 32];
    }
  }
  // Per-lane C-row constants: C vgpr v -> M = i0 + v + 8*half
  float ra1[8], rs1[8];
#pragma unroll
  for (int v = 0; v < 8; ++v) {
    const int m = i0 + v + 8 * half;
    ra1[v] = attn1[m];
    rs1[v] = sq1b[m];
  }

  float acc = 0.f;
  for (int jt = wave; jt < SS / 16; jt += 8) {   // uniform per wave: EXEC all-ones
    const int j0 = jt * 16;
    const int n  = j0 + nIdx;
    // B fragments (32x16 f16): lane N = lane%16, element e -> K = 16*half + e
    v16h b_lo, b_hi;
    const float4* row = (const float4*)(v2b + (size_t)n * VDIM);
#pragma unroll
    for (int q = 0; q < 4; ++q) {
      float4 f = row[4 * half + q];
      b_lo[4*q+0] = (_Float16)f.x; b_lo[4*q+1] = (_Float16)f.y;
      b_lo[4*q+2] = (_Float16)f.z; b_lo[4*q+3] = (_Float16)f.w;
      float4 g = row[8 + 4 * half + q];
      b_hi[4*q+0] = (_Float16)g.x; b_hi[4*q+1] = (_Float16)g.y;
      b_hi[4*q+2] = (_Float16)g.z; b_hi[4*q+3] = (_Float16)g.w;
    }
    v8f c = {};
    c = __builtin_amdgcn_wmma_f32_16x16x32_f16(false, a_lo, false, b_lo,
                                               (short)0, c, false, false);
    c = __builtin_amdgcn_wmma_f32_16x16x32_f16(false, a_hi, false, b_hi,
                                               (short)0, c, false, false);
    const float ra2 = attn2[n];
    const float rs2 = sq2b[n];
#pragma unroll
    for (int v = 0; v < 8; ++v) {
      float d2 = fmaxf(rs1[v] + rs2 - 2.f * c[v], 0.f);
      acc += ra1[v] * ra2 * __expf(-0.5f * d2);   // GAMMA = 0.5
    }
  }

  __shared__ float red[256];
  red[tid] = acc;
  __syncthreads();
  for (int off = 128; off > 0; off >>= 1) {
    if (tid < off) red[tid] += red[tid + off];
    __syncthreads();
  }
  if (tid == 0) partials[blockIdx.x] = red[0];
}

// ---- Kernel 5: deterministic final reduction + loss ----
__global__ void finalize_kernel(const float* __restrict__ partials,
                                float* __restrict__ out_loss) {
  __shared__ float red[256];
  const int t = threadIdx.x;
  float s = partials[t] + partials[t + 256] + partials[t + 512] + partials[t + 768];
  red[t] = s; __syncthreads();
  for (int off = 128; off > 0; off >>= 1) {
    if (t < off) red[t] += red[t + off];
    __syncthreads();
  }
  if (t == 0) {
    const float R = red[0];
    // sum(w_attn) == B exactly (softmax rows sum to 1) -> penalty = lambda*(B - R)
    const float P = 0.1f * ((float)BB - R);
    const float sim = R - P + fmaxf(R, fabsf(P));
    out_loss[0] = 1.f / (sim * sim + 1e-6f);
  }
}

extern "C" void kernel_launch(void* const* d_in, const int* in_sizes, int n_in,
                              void* d_out, int out_size, void* d_ws, size_t ws_size,
                              hipStream_t stream) {
  (void)in_sizes; (void)n_in; (void)out_size; (void)ws_size;
  const float* query1  = (const float*)d_in[0];
  const float* query2  = (const float*)d_in[1];
  const float* keys1   = (const float*)d_in[2];
  const float* keys2   = (const float*)d_in[3];
  const float* values1 = (const float*)d_in[4];
  const float* values2 = (const float*)d_in[5];
  const float* W       = (const float*)d_in[6];
  float* out = (float*)d_out;   // [attn1 (B*S) | attn2 (B*S) | loss (1)]

  // Workspace layout (floats)
  float* ws       = (float*)d_ws;
  float* tq       = ws;                      // 2*B*KD   = 1024
  float* scores1  = ws + 1024;               // B*S      = 16384
  float* scores2  = scores1 + BB * SS;       // 16384
  float* sq1      = scores2 + BB * SS;       // 16384
  float* sq2      = sq1 + BB * SS;           // 16384
  float* partials = sq2 + BB * SS;           // 1024

  proj_norm_kernel<<<2 * BB, KDIM, 0, stream>>>(query1, query2, W, tq);
  scores_sq_kernel<<<(BB * SS) / 256, 256, 0, stream>>>(
      keys1, keys2, values1, values2, tq, scores1, scores2, sq1, sq2);
  softmax_kernel<<<2 * BB, 256, 0, stream>>>(scores1, scores2, out);
  rbf_wmma_kernel<<<BB * (SS / 16), 256, 0, stream>>>(
      values1, values2, sq1, sq2, out, partials);
  finalize_kernel<<<1, 256, 0, stream>>>(partials, out + 2 * BB * SS);
}
